// MTLSTM_9311489098197
// MI455X (gfx1250) — matrix operand order repaired and checked by reading, not compiled, not run on hardware
//
#include <hip/hip_runtime.h>
#include <hip/hip_bf16.h>

typedef __bf16 bf16_t;
typedef __attribute__((ext_vector_type(16))) __bf16 v16bf;
typedef __attribute__((ext_vector_type(8)))  __bf16 v8bf;
typedef __attribute__((ext_vector_type(8)))  float  v8f;

#define BATCH 256
#define SEQ_T 2048
#define NIN   8
#define HID   256
#define MTILE 16
#define NWG   (BATCH / MTILE)   // 16 persistent workgroups
#define NTHREADS 1024
#define NWAVES 32
// N-tiles per step: 48 gate tiles (cols 256..1023 of 4H space: f,g,o; i-gate is dead)
// + 128 decomp tiles (8 subspaces x 256)
#define NGTILES 48
#define NDTILES 128
#define NTILES  (NGTILES + NDTILES)

// ---- fast activations: keep the 2048-step critical path on TRANS/VALU ops ----
__device__ __forceinline__ float fast_rcp(float x) {
#if __has_builtin(__builtin_amdgcn_rcpf)
  return __builtin_amdgcn_rcpf(x);
#else
  return 1.0f / x;
#endif
}

__device__ __forceinline__ float fast_tanh(float x) {
#if __has_builtin(__builtin_amdgcn_tanh_f32)
  return __builtin_amdgcn_tanh_f32(x);          // V_TANH_F32 (gfx1250 TRANS op)
#else
  // tanh(x) = 1 - 2/(exp(2x)+1); v_exp_f32 based, branch-free
  float e = __expf(2.0f * x);
  return 1.0f - 2.0f * fast_rcp(e + 1.0f);
#endif
}

__device__ __forceinline__ float fast_sigmoid(float x) {
  return fast_rcp(1.0f + __expf(-x));
}

__global__ void cvt_f32_to_bf16(const float* __restrict__ src,
                                bf16_t* __restrict__ dst, int n) {
  int i = blockIdx.x * blockDim.x + threadIdx.x;
  if (i < n) dst[i] = (bf16_t)src[i];
}

__global__ __launch_bounds__(NTHREADS) void mtlstm_persistent(
    const float* __restrict__ xd, const float* __restrict__ dtm,
    const float* __restrict__ w_ih, const float* __restrict__ bias,
    const float* __restrict__ b_dec,
    const bf16_t* __restrict__ whh_bf, const bf16_t* __restrict__ wdec_bf,
    const float* __restrict__ w_lin, const float* __restrict__ b_lin,
    float* __restrict__ out)
{
  extern __shared__ char smem_raw[];
  float* sC32  = (float*)smem_raw;            // [16][256] fp32 cell state
  float* sF    = sC32  + MTILE*HID;           // [16][256] f gate (activated)
  float* sG    = sF    + MTILE*HID;           // [16][256] g gate
  float* sO    = sG    + MTILE*HID;           // [16][256] o gate
  float* sSum  = sO    + MTILE*HID;           // [16][256] sum_k cs
  float* sSumD = sSum  + MTILE*HID;           // [16][256] sum_k cs*decay
  float* sWih  = sSumD + MTILE*HID;           // [1024][8] W_ih
  float* sBias = sWih  + 4*HID*NIN;           // [1024]
  float* sBdec = sBias + 4*HID;               // [8][256]
  float* sX    = sBdec + NIN*HID;             // [16][8] x_t
  float* sDec  = sX    + MTILE*NIN;           // [16][8] decay_t
  bf16_t* sHbf = (bf16_t*)(sDec + MTILE*NIN); // [16][256] hidden (bf16 A-frag)
  bf16_t* sCbf = sHbf + MTILE*HID;            // [16][256] cell   (bf16 A-frag)

  const int tid  = threadIdx.x;
  const int lane = tid & 31;
  const int wave = tid >> 5;
  const int kh   = lane >> 4;   // wave half (selects K sub-chunks / M rows 0-7 vs 8-15)
  const int nl   = lane & 15;   // A row / B column within 16x16 tile
  const int b0   = blockIdx.x * MTILE;

  // ---- one-time init ----
  for (int i = tid; i < 4*HID*NIN; i += NTHREADS) sWih[i]  = w_ih[i];
  for (int i = tid; i < 4*HID;     i += NTHREADS) sBias[i] = bias[i];
  for (int i = tid; i < NIN*HID;   i += NTHREADS) sBdec[i] = b_dec[i];
  for (int i = tid; i < MTILE*HID; i += NTHREADS) {
    sC32[i] = 0.f; sSum[i] = 0.f; sSumD[i] = 0.f;
    sHbf[i] = (bf16_t)0.f; sCbf[i] = (bf16_t)0.f;
  }
  if (tid < MTILE*NIN) {
    int m = tid >> 3, k = tid & 7;
    size_t idx = ((size_t)(b0 + m) * SEQ_T) * NIN + k;
    sX[tid]  = xd[idx];
    sDec[tid] = fast_rcp(__logf(2.718281828459045f + dtm[idx]));
  }
  __syncthreads();

  for (int t = 0; t < SEQ_T; ++t) {
    // ---- phase 1: per-step GEMM via v_wmma_f32_16x16x32_bf16 ----
    for (int tt = wave; tt < NTILES; tt += NWAVES) {   // wave-uniform: EXEC all-1s at WMMA
      const bool isGate = (tt < NGTILES);
      const bf16_t* Bp; const bf16_t* Ap;
      float biasv; int kidx = 0, j = 0, col = 0;
      if (isGate) {
        col = 256 + tt * 16 + nl;                      // f/g/o columns only
        Bp = whh_bf + (size_t)col * HID;               // W_hh row (K-contiguous)
        Ap = sHbf;  biasv = sBias[col];
      } else {
        int d = (tt - NGTILES) * 16 + nl;              // 0..2047
        kidx = d >> 8; j = d & 255;
        Bp = wdec_bf + (size_t)d * HID;                // W_decomp row (K-contiguous)
        Ap = sCbf;  biasv = sBdec[kidx * HID + j];
      }
      v8f acc = { biasv, biasv, biasv, biasv, biasv, biasv, biasv, biasv };
      // A frag (16x32 bf16): lane holds row nl, K chunks {kb+8*kh .. +7} and {+16}
      const bf16_t* Arow = Ap + nl * HID + kh * 8;
      // B frag (32x16 bf16): lane holds column nl, K chunks {kb+8*kh .. +7} and {+16}
      const bf16_t* Brow = Bp + kh * 8;
      #pragma unroll
      for (int kb = 0; kb < HID; kb += 32) {
        union { v16bf v; v8bf h[2]; } a, b;
        a.h[0] = *(const v8bf*)(Arow + kb);
        a.h[1] = *(const v8bf*)(Arow + kb + 16);
        b.h[0] = *(const v8bf*)(Brow + kb);
        b.h[1] = *(const v8bf*)(Brow + kb + 16);
        acc = __builtin_amdgcn_wmma_f32_16x16x32_bf16(false, a.v, false, b.v,
                                                      (short)0, acc, false, false);
      }
      // D layout: acc[i] = D[row = i + 8*kh][col = nl]
      if (isGate) {
        #pragma unroll
        for (int i = 0; i < 8; ++i) {                  // + x_t @ W_ih.T (rank-8, VALU)
          int m = i + kh * 8;
          float s = 0.f;
          #pragma unroll
          for (int k = 0; k < NIN; ++k) s += sX[m*NIN + k] * sWih[col*NIN + k];
          acc[i] += s;
        }
        const int gsel = col >> 8;                     // 1=f, 2=g, 3=o (tile-uniform)
        const int jj = col & 255;
        #pragma unroll
        for (int i = 0; i < 8; ++i) {
          int m = i + kh * 8;
          float v = acc[i];
          if (gsel == 2) sG[m*HID + jj] = fast_tanh(v);
          else           (gsel == 1 ? sF : sO)[m*HID + jj] = fast_sigmoid(v);
        }
      } else {
        #pragma unroll
        for (int i = 0; i < 8; ++i) {
          int m = i + kh * 8;
          float v = fast_tanh(acc[i]);                 // cs[m, kidx, j]
          atomicAdd(&sSum [m*HID + j], v);             // ds_add_f32 reduction over k
          atomicAdd(&sSumD[m*HID + j], v * sDec[m*NIN + kidx]);
        }
      }
    }
    __syncthreads();
    // ---- phase 2: state update (faithful bug: + (IN-1)*g) ----
    for (int e = tid; e < MTILE*HID; e += NTHREADS) {
      float f = sF[e], g = sG[e], o = sO[e];
      float c_star = sC32[e] - sSum[e] + sSumD[e];
      float cn = f * c_star + (float)(NIN - 1) * g;
      sC32[e] = cn;
      sCbf[e] = (bf16_t)cn;
      sHbf[e] = (bf16_t)(o * fast_tanh(cn));
      sSum[e] = 0.f; sSumD[e] = 0.f;                   // re-zero for next step
    }
    if (t + 1 < SEQ_T && tid < MTILE*NIN) {            // prefetch x_{t+1}, decay_{t+1}
      int m = tid >> 3, k = tid & 7;
      size_t idx = ((size_t)(b0 + m) * SEQ_T + (t + 1)) * NIN + k;
      sX[tid]  = xd[idx];
      sDec[tid] = fast_rcp(__logf(2.718281828459045f + dtm[idx]));
    }
    __syncthreads();
  }

  // ---- final: relu(w_lin @ h_final + b_lin), OUT=1 ----
  if (tid < MTILE) {
    float s = 0.f;
    for (int jx = 0; jx < HID; ++jx) s += (float)sHbf[tid*HID + jx] * w_lin[jx];
    s += b_lin[0];
    out[b0 + tid] = s > 0.f ? s : 0.f;
  }
}

extern "C" void kernel_launch(void* const* d_in, const int* in_sizes, int n_in,
                              void* d_out, int out_size, void* d_ws, size_t ws_size,
                              hipStream_t stream) {
  const float* xd    = (const float*)d_in[0];
  const float* dtm   = (const float*)d_in[1];
  const float* w_ih  = (const float*)d_in[2];
  const float* w_hh  = (const float*)d_in[3];
  const float* bias  = (const float*)d_in[4];
  const float* w_dec = (const float*)d_in[5];
  const float* b_dec = (const float*)d_in[6];
  const float* w_lin = (const float*)d_in[7];
  const float* b_lin = (const float*)d_in[8];
  float* out = (float*)d_out;

  const int n_whh  = 4*HID*HID;    // 262144
  const int n_wdec = NIN*HID*HID;  // 524288
  bf16_t* whh_bf  = (bf16_t*)d_ws;                 // [1024][256] bf16
  bf16_t* wdec_bf = whh_bf + n_whh;                // [2048][256] bf16 (k*H+j major)

  cvt_f32_to_bf16<<<(n_whh  + 255)/256, 256, 0, stream>>>(w_hh,  whh_bf,  n_whh);
  cvt_f32_to_bf16<<<(n_wdec + 255)/256, 256, 0, stream>>>(w_dec, wdec_bf, n_wdec);

  size_t smem = (size_t)(6*MTILE*HID + 4*HID*NIN + 4*HID + NIN*HID + 2*MTILE*NIN)
                    * sizeof(float)
              + (size_t)(2*MTILE*HID) * sizeof(bf16_t);   // ~157 KB < 320 KB/WGP
  mtlstm_persistent<<<NWG, NTHREADS, smem, stream>>>(
      xd, dtm, w_ih, bias, b_dec, whh_bf, wdec_bf, w_lin, b_lin, out);
}
